// HybridGATLSTM_5660766896649
// MI455X (gfx1250) — matrix operand order repaired and checked by reading, not compiled, not run on hardware
//
#include <hip/hip_runtime.h>
#include <hip/hip_bf16.h>
#include <cstdint>

// ---------------------------------------------------------------------------
// B=16, S=96, N=128, H=128, F1=32, F2=16, COMB=2176
// ---------------------------------------------------------------------------

typedef __attribute__((ext_vector_type(16))) _Float16 v16h;
typedef __attribute__((ext_vector_type(8)))  float    v8f;

#define DEV __device__ __forceinline__

DEV v8f wmma_f16(v16h a, v16h b, v8f c) {
  return __builtin_amdgcn_wmma_f32_16x16x32_f16(
      false, a, false, b, (short)0, c, false, false);
}

// A-matrix (16x32 f16), row-major source: contiguous per-lane K-runs -> b128s.
DEV v16h load_a_16x32(const _Float16* A, int lda, int m0, int k0) {
  int lane = threadIdx.x & 31;
  int m  = m0 + (lane & 15);
  int hf = lane >> 4;
  const _Float16* row = A + m * lda + k0;
  v16h a;
#pragma unroll
  for (int i = 0; i < 4; ++i) {
    int k = hf * 16 + 2 * i;
    a[2 * i]     = row[k];
    a[2 * i + 1] = row[k + 1];
  }
#pragma unroll
  for (int i = 0; i < 4; ++i) {
    int k = hf * 16 + 8 + 2 * i;
    a[8 + 2 * i]     = row[k];
    a[8 + 2 * i + 1] = row[k + 1];
  }
  return a;
}

// ---------------------------------------------------------------------------
// Swizzled-B storage. For a KxN matrix consumed as WMMA B, lane `l` of tile
// (kc, nt) needs B[kc*32 + 16*(l>>4) + e][nt*16 + (l&15)], e = 0..15 — a
// contiguous K-run. Store as [kc*NT+nt][lane][16] so consumption is one
// aligned 32-byte LDS load (2x ds_load_b128), no per-element gather.
// ---------------------------------------------------------------------------
DEV v16h load_b_sw(const _Float16* Bsw, int ntiles, int kc, int nt) {
  int lane = threadIdx.x & 31;
  const _Float16* p = Bsw + (size_t)(((kc * ntiles + nt) * 32) + lane) * 16;
  return *(const v16h*)p;
}

// swizzled destination index for element (k, n) of a KxN B-matrix
DEV int bsw_index(int k, int n, int ntiles) {
  int kc = k >> 5;
  int hf = (k >> 4) & 1;
  int e  = k & 15;
  int lane = (n & 15) + 16 * hf;
  return (((kc * ntiles + (n >> 4)) * 32) + lane) * 16 + e;
}

// B-tile gather from f32 global (no reuse -> gather once is optimal)
DEV v16h load_b_32x16_g(const float* B, int ldb, int k0, int n0) {
  int lane = threadIdx.x & 31;
  int n  = n0 + (lane & 15);
  int hf = lane >> 4;
  v16h b;
#pragma unroll
  for (int i = 0; i < 8; ++i) {
    int k = k0 + 16 * hf + 2 * i;
    b[2 * i]     = (_Float16)B[(size_t)k * ldb + n];
    b[2 * i + 1] = (_Float16)B[(size_t)(k + 1) * ldb + n];
  }
  return b;
}

// C/D (16x16 f32): VGPR j -> row m0 + 8*(lane>>4) + j, col n0 + (lane&15).
DEV void store_d_16x16(float* C, int ldc, int m0, int n0, v8f d) {
  int lane = threadIdx.x & 31;
  int n  = n0 + (lane & 15);
  int hf = lane >> 4;
#pragma unroll
  for (int j = 0; j < 8; ++j)
    C[(size_t)(m0 + hf * 8 + j) * ldc + n] = d[j];
}

// ---------------------------------------------------------------------------
// Kernel 1: fused x_emb + GAT1 + GAT2 per graph (1536 blocks, 256 threads)
// ---------------------------------------------------------------------------
#define GAT_LDS 109952

__global__ void __launch_bounds__(256)
gat_fused_kernel(const float* __restrict__ x, const float* __restrict__ adj,
                 const float* __restrict__ Wemb, const float* __restrict__ bemb,
                 const float* __restrict__ W1, const float* __restrict__ a1,
                 const float* __restrict__ W2, const float* __restrict__ a2,
                 float* __restrict__ attn2_out, float* __restrict__ xg_out) {
  extern __shared__ char smem[];
  _Float16* sXemb  = (_Float16*)(smem + 0);       // 128x32 A row-major  8192B
  _Float16* sW1sw  = (_Float16*)(smem + 8192);    // 32x32 B swizzled   2048B
  _Float16* sW2sw  = (_Float16*)(smem + 10240);   // 32x16 B swizzled   1024B
  _Float16* sWh1sw = (_Float16*)(smem + 11264);   // 128x32 B swizzled  8192B
  _Float16* sXg1h  = (_Float16*)(smem + 19456);   // 128x32 A row-major 8192B
  _Float16* sWh2sw = (_Float16*)(smem + 27648);   // 128x16 B swizzled  4096B
  _Float16* sAttn  = (_Float16*)(smem + 31744);   // 128x128 A         32768B
  float*    sWh1   = (float*)(smem + 64512);      // 128x32            16384B
  float*    sXg1   = (float*)(smem + 80896);      // 128x32            16384B
  float*    sWh2   = (float*)(smem + 97280);      // 128x16             8192B
  float*    sLR    = (float*)(smem + 105472);     // 512                2048B
  float*    sA     = (float*)(smem + 107520);     // 96                  384B
  unsigned* sAdj   = (unsigned*)(smem + 107904);  // 512                2048B

  const int g    = blockIdx.x;   // graph = b*S + s
  const int s    = g % 96;
  const int tid  = threadIdx.x;
  const int wave = tid >> 5;
  const int lane = tid & 31;

  // --- parameters (B matrices pre-swizzled) & adjacency bitmask ---
  for (int d = tid; d < 1024; d += 256) {          // W1: K=32, NT=2
    int e = d & 15, ln = (d >> 4) & 31, nt = (d >> 9) & 1;
    int k = 16 * (ln >> 4) + e;
    int n = nt * 16 + (ln & 15);
    sW1sw[d] = (_Float16)W1[k * 32 + n];
  }
  for (int d = tid; d < 512; d += 256) {           // W2: K=32, NT=1
    int e = d & 15, ln = (d >> 4) & 31;
    int k = 16 * (ln >> 4) + e;
    int n = ln & 15;
    sW2sw[d] = (_Float16)W2[k * 16 + n];
  }
  if (tid < 64) sA[tid] = a1[tid];
  else if (tid < 96) sA[tid] = a2[tid - 64];
  for (int i = tid; i < 512; i += 256) {
    int n = i >> 2, w = i & 3;
    unsigned bits = 0;
    const float* arow = adj + n * 128 + w * 32;
#pragma unroll
    for (int bp = 0; bp < 32; ++bp)
      bits |= (arow[bp] > 0.0f) ? (1u << bp) : 0u;
    sAdj[i] = bits;
  }
  // --- x_emb (FI=1: outer product) ---
  const float* xrow = x + (size_t)g * 128;
  for (int i = tid; i < 128 * 32; i += 256) {
    int n = i >> 5, f = i & 31;
    sXemb[i] = (_Float16)(xrow[n] * Wemb[f] + bemb[f]);
  }
  __syncthreads();

  // --- Wh1 = Xemb(128x32) @ W1(32x32): 8x2 tiles, 2/wave ---
#pragma unroll
  for (int i = 0; i < 2; ++i) {
    int id = wave * 2 + i;
    int m0 = (id >> 1) * 16, n0 = (id & 1) * 16;
    v16h av = load_a_16x32(sXemb, 32, m0, 0);
    v16h bv = load_b_sw(sW1sw, 2, 0, n0 >> 4);
    v8f acc = {};
    acc = wmma_f16(av, bv, acc);
    int n = n0 + (lane & 15), hf = lane >> 4;
#pragma unroll
    for (int j = 0; j < 8; ++j) {
      int m = m0 + hf * 8 + j;
      sWh1[m * 32 + n] = acc[j];
      sWh1sw[bsw_index(m, n, 2)] = (_Float16)acc[j];  // Wh1 is B of out1 (K=128)
    }
  }
  __syncthreads();

  // --- left/right projections (attention 1) ---
  if (tid < 128) {
    float l = 0.f, r = 0.f;
    const float* wr = sWh1 + tid * 32;
#pragma unroll
    for (int f = 0; f < 32; ++f) { l += wr[f] * sA[f]; r += wr[f] * sA[32 + f]; }
    sLR[tid] = l; sLR[128 + tid] = r;
  }
  __syncthreads();

  // --- attn1: leaky_relu + mask + row softmax (wave-per-16-rows) ---
  for (int r = 0; r < 16; ++r) {
    int n = wave * 16 + r;
    float lv = sLR[n];
    float e[4];
#pragma unroll
    for (int t = 0; t < 4; ++t) {
      float v = lv + sLR[128 + t * 32 + lane];
      v = v > 0.f ? v : 0.2f * v;
      bool on = (sAdj[n * 4 + t] >> lane) & 1u;
      e[t] = on ? v : -1e30f;
    }
    float mx = fmaxf(fmaxf(e[0], e[1]), fmaxf(e[2], e[3]));
#pragma unroll
    for (int off = 16; off > 0; off >>= 1) mx = fmaxf(mx, __shfl_xor(mx, off, 32));
    float sum = 0.f;
#pragma unroll
    for (int t = 0; t < 4; ++t) { e[t] = __expf(e[t] - mx); sum += e[t]; }
#pragma unroll
    for (int off = 16; off > 0; off >>= 1) sum += __shfl_xor(sum, off, 32);
    float inv = 1.0f / sum;
#pragma unroll
    for (int t = 0; t < 4; ++t)
      sAttn[n * 128 + t * 32 + lane] = (_Float16)(e[t] * inv);
  }
  __syncthreads();

  // --- out1 = elu(attn1 @ Wh1): 8x2 tiles x 4 K-chunks ---
#pragma unroll
  for (int i = 0; i < 2; ++i) {
    int id = wave * 2 + i;
    int m0 = (id >> 1) * 16, n0 = (id & 1) * 16;
    v8f acc = {};
#pragma unroll
    for (int kc = 0; kc < 4; ++kc) {
      v16h av = load_a_16x32(sAttn, 128, m0, kc * 32);
      v16h bv = load_b_sw(sWh1sw, 2, kc, n0 >> 4);
      acc = wmma_f16(av, bv, acc);
    }
    int n = n0 + (lane & 15), hf = lane >> 4;
#pragma unroll
    for (int j = 0; j < 8; ++j) {
      float v = acc[j];
      v = v > 0.f ? v : (__expf(v) - 1.0f);  // elu
      int m = m0 + hf * 8 + j;
      sXg1[m * 32 + n]  = v;
      sXg1h[m * 32 + n] = (_Float16)v;       // A of Wh2 GEMM (row-major)
    }
  }
  __syncthreads();

  // --- Wh2 = Xg1(128x32) @ W2(32x16): 8 tiles, 1/wave ---
  {
    int m0 = wave * 16;
    v16h av = load_a_16x32(sXg1h, 32, m0, 0);
    v16h bv = load_b_sw(sW2sw, 1, 0, 0);
    v8f acc = {};
    acc = wmma_f16(av, bv, acc);
    int n = lane & 15, hf = lane >> 4;
#pragma unroll
    for (int j = 0; j < 8; ++j) {
      int m = m0 + hf * 8 + j;
      sWh2[m * 16 + n] = acc[j];
      sWh2sw[bsw_index(m, n, 1)] = (_Float16)acc[j];  // Wh2 is B of out2 (K=128)
    }
  }
  __syncthreads();

  if (tid < 128) {
    float l = 0.f, r = 0.f;
    const float* wr = sWh2 + tid * 16;
#pragma unroll
    for (int f = 0; f < 16; ++f) { l += wr[f] * sA[64 + f]; r += wr[f] * sA[80 + f]; }
    sLR[256 + tid] = l; sLR[384 + tid] = r;
  }
  __syncthreads();

  // --- attn2: softmax + stream to HBM (bulk output, coalesced) ---
  float* gattn = attn2_out + (size_t)g * 16384;
  for (int r = 0; r < 16; ++r) {
    int n = wave * 16 + r;
    float lv = sLR[256 + n];
    float e[4];
#pragma unroll
    for (int t = 0; t < 4; ++t) {
      float v = lv + sLR[384 + t * 32 + lane];
      v = v > 0.f ? v : 0.2f * v;
      bool on = (sAdj[n * 4 + t] >> lane) & 1u;
      e[t] = on ? v : -1e30f;
    }
    float mx = fmaxf(fmaxf(e[0], e[1]), fmaxf(e[2], e[3]));
#pragma unroll
    for (int off = 16; off > 0; off >>= 1) mx = fmaxf(mx, __shfl_xor(mx, off, 32));
    float sum = 0.f;
#pragma unroll
    for (int t = 0; t < 4; ++t) { e[t] = __expf(e[t] - mx); sum += e[t]; }
#pragma unroll
    for (int off = 16; off > 0; off >>= 1) sum += __shfl_xor(sum, off, 32);
    float inv = 1.0f / sum;
#pragma unroll
    for (int t = 0; t < 4; ++t) {
      float av = e[t] * inv;
      sAttn[n * 128 + t * 32 + lane] = (_Float16)av;
      gattn[(size_t)n * 128 + t * 32 + lane] = av;
    }
  }

  // --- out2 only for last timestep of each batch ---
  if (s == 95) {
    __syncthreads();
    int m0 = wave * 16;
    v8f acc = {};
#pragma unroll
    for (int kc = 0; kc < 4; ++kc) {
      v16h av = load_a_16x32(sAttn, 128, m0, kc * 32);
      v16h bv = load_b_sw(sWh2sw, 1, kc, 0);
      acc = wmma_f16(av, bv, acc);
    }
    int bidx = g / 96;
    store_d_16x16(xg_out + (size_t)bidx * 2048, 16, m0, 0, acc);
  }
}

// ---------------------------------------------------------------------------
// Kernel 2: 3-layer LSTM, persistent single workgroup.
// [Wih;Whh]^T (256x512 f16 = 256KB) lives in LDS, pre-swizzled into WMMA-B
// register layout (swizzle paid once per layer, consumed 96x with b128 loads).
// Per-step input via CDNA5 async global->LDS copy.
// ---------------------------------------------------------------------------
#define LSTM_LDS 321536

__global__ void __launch_bounds__(256)
lstm_kernel(const float* __restrict__ x,
            const float* __restrict__ Wih, const float* __restrict__ Whh,
            const float* __restrict__ bih, const float* __restrict__ bhh,
            float* __restrict__ seqA, float* __restrict__ seqB,
            float* __restrict__ xl_out) {
  extern __shared__ char smem[];
  _Float16* sWsw = (_Float16*)(smem);          // 256x512 B swizzled 262144B
  _Float16* sZ   = (_Float16*)(smem + 262144); // 16x256 A row-major  8192B
  float*    sG   = (float*)(smem + 270336);    // 16x512             32768B
  float*    sC   = (float*)(smem + 303104);    // 16x128              8192B
  float*    sB   = (float*)(smem + 311296);    // 512                 2048B
  float*    sX   = (float*)(smem + 313344);    // 16x128 async stage  8192B

  const int tid  = threadIdx.x;
  const int wave = tid >> 5;

  for (int l = 0; l < 3; ++l) {
    const float* wi = Wih + (size_t)l * 512 * 128;
    const float* wh = Whh + (size_t)l * 512 * 128;
    // weights -> LDS f16, directly in swizzled B layout (KC=8, NT=32)
    for (int d = tid; d < 256 * 512; d += 256) {
      int e = d & 15, ln = (d >> 4) & 31;
      int tile = d >> 9;                 // 0..255
      int kc = tile >> 5, nt = tile & 31;
      int n = nt * 16 + (ln & 15);
      int k = kc * 32 + 16 * (ln >> 4) + e;
      float w = (k < 128) ? wi[n * 128 + k] : wh[n * 128 + (k - 128)];
      sWsw[d] = (_Float16)w;
    }
    for (int i = tid; i < 512; i += 256) sB[i] = bih[l * 512 + i] + bhh[l * 512 + i];
    for (int i = tid; i < 2048; i += 256) {
      sC[i] = 0.f;
      int b = i >> 7, h = i & 127;
      sZ[b * 256 + 128 + h] = (_Float16)0.f;   // h0 = 0
    }
    const float* src = (l == 0) ? x : ((l == 1) ? seqA : seqB);
    float* dst = (l == 0) ? seqA : seqB;
    __syncthreads();

    for (int t = 0; t < 96; ++t) {
      // ---- async copy x_t (16x128 f32) global -> LDS staging ----
      {
        unsigned sbase = (unsigned)(uintptr_t)sX;
#pragma unroll
        for (int r = 0; r < 2; ++r) {
          int li = tid * 2 + r;           // 0..511 (16B chunks)
          int b = li >> 5, ch = li & 31;
          unsigned long long gsrc =
              (unsigned long long)(uintptr_t)(src + (size_t)(b * 96 + t) * 128 + ch * 4);
          unsigned ldst = sbase + (unsigned)(b * 512 + ch * 16);
          asm volatile("global_load_async_to_lds_b128 %0, %1, off"
                       :: "v"(ldst), "v"(gsrc) : "memory");
        }
        asm volatile("s_wait_asynccnt 0" ::: "memory");
      }
      if (t < 95 && tid < 16)
        __builtin_prefetch(src + (size_t)(tid * 96 + t + 1) * 128, 0, 1);
      __syncthreads();
      for (int i = tid; i < 2048; i += 256) {
        int b = i >> 7, n = i & 127;
        sZ[b * 256 + n] = (_Float16)sX[i];
      }
      __syncthreads();
      // ---- g = z(16x256) @ W(256x512): 32 N-tiles, 8 K-chunks ----
#pragma unroll
      for (int jt = 0; jt < 4; ++jt) {
        int nt = wave * 4 + jt;
        v8f acc = {};
#pragma unroll
        for (int kc = 0; kc < 8; ++kc) {
          v16h av = load_a_16x32(sZ, 256, 0, kc * 32);
          v16h bv = load_b_sw(sWsw, 32, kc, nt);
          acc = wmma_f16(av, bv, acc);
        }
        store_d_16x16(sG, 512, 0, nt * 16, acc);
      }
      __syncthreads();
      // ---- gates ----
      for (int i = tid; i < 2048; i += 256) {
        int b = i >> 7, h = i & 127;
        const float* gr = sG + b * 512;
        float ig = 1.f / (1.f + __expf(-(gr[h]       + sB[h])));
        float fg = 1.f / (1.f + __expf(-(gr[128 + h] + sB[128 + h])));
        float gg = tanhf(gr[256 + h] + sB[256 + h]);
        float og = 1.f / (1.f + __expf(-(gr[384 + h] + sB[384 + h])));
        float c  = fg * sC[i] + ig * gg;
        float hv = og * tanhf(c);
        sC[i] = c;
        sZ[b * 256 + 128 + h] = (_Float16)hv;
        if (l < 2)        dst[(size_t)(b * 96 + t) * 128 + h] = hv;
        else if (t == 95) xl_out[b * 128 + h] = hv;
      }
      __syncthreads();
    }
    __syncthreads();
  }
}

// ---------------------------------------------------------------------------
// Kernel 3: heads. grid=3 (k), 128 threads. K=2176 -> 68 WMMA chunks.
// ---------------------------------------------------------------------------
#define HEAD_LDS 73728

__global__ void __launch_bounds__(128)
head_kernel(const float* __restrict__ xl_out, const float* __restrict__ xg_out,
            const float* __restrict__ hW1, const float* __restrict__ hb1,
            const float* __restrict__ hW2, const float* __restrict__ hb2,
            const float* __restrict__ W3d, const float* __restrict__ b3d,
            const float* __restrict__ W3r, const float* __restrict__ b3r,
            const float* __restrict__ W3v, const float* __restrict__ b3v,
            float* __restrict__ out) {
  extern __shared__ char smem[];
  _Float16* sC  = (_Float16*)smem;             // 16x2176  69632B
  _Float16* sH1 = (_Float16*)(smem + 69632);   // 16x64     2048B
  float*    sH2 = (float*)(smem + 71680);      // 16x32     2048B

  const int k = blockIdx.x, tid = threadIdx.x, wave = tid >> 5, lane = tid & 31;

  for (int i = tid; i < 16 * 2176; i += 128) {
    int b = i / 2176, c = i % 2176;
    float v = (c < 128) ? xl_out[b * 128 + c] : xg_out[(size_t)b * 2048 + (c - 128)];
    sC[i] = (_Float16)v;
  }
  __syncthreads();

  // h1 = relu(c @ W1[k] + b1[k]) : 4 N-tiles (1/wave), 68 K-chunks
  {
    const float* Bk = hW1 + (size_t)k * 2176 * 64;
    int n0 = wave * 16;
    v8f acc = {};
    for (int kc = 0; kc < 68; ++kc) {
      if (kc + 1 < 68) __builtin_prefetch(Bk + (size_t)(kc + 1) * 32 * 64, 0, 1);
      v16h av = load_a_16x32(sC, 2176, 0, kc * 32);
      v16h bv = load_b_32x16_g(Bk, 64, kc * 32, n0);
      acc = wmma_f16(av, bv, acc);
    }
    int n = n0 + (lane & 15), hf = lane >> 4;
#pragma unroll
    for (int j = 0; j < 8; ++j) {
      int m = hf * 8 + j;
      float v = acc[j] + hb1[k * 64 + n];
      sH1[m * 64 + n] = (_Float16)(v > 0.f ? v : 0.f);
    }
  }
  __syncthreads();

  // h2 = relu(h1 @ W2[k] + b2[k]) : 2 N-tiles (waves 0,1), 2 K-chunks
  if (wave < 2) {
    const float* Bk = hW2 + (size_t)k * 64 * 32;
    int n0 = wave * 16;
    v8f acc = {};
#pragma unroll
    for (int kc = 0; kc < 2; ++kc) {
      v16h av = load_a_16x32(sH1, 64, 0, kc * 32);
      v16h bv = load_b_32x16_g(Bk, 32, kc * 32, n0);
      acc = wmma_f16(av, bv, acc);
    }
    int n = n0 + (lane & 15), hf = lane >> 4;
#pragma unroll
    for (int j = 0; j < 8; ++j) {
      int m = hf * 8 + j;
      float v = acc[j] + hb2[k * 32 + n];
      sH2[m * 32 + n] = v > 0.f ? v : 0.f;
    }
  }
  __syncthreads();

  // final projections -> out[0:16]=d, out[16:32]=r, out[32:64]=v
  if (k == 0 && tid < 16) {
    float acc = b3d[0];
    for (int e = 0; e < 32; ++e) acc += sH2[tid * 32 + e] * W3d[e];
    out[tid] = acc;
  } else if (k == 1 && tid < 16) {
    float acc = b3r[0];
    for (int e = 0; e < 32; ++e) acc += sH2[tid * 32 + e] * W3r[e];
    out[16 + tid] = acc;
  } else if (k == 2 && tid < 32) {
    int b = tid >> 1, j = tid & 1;
    float acc = b3v[j];
    for (int e = 0; e < 32; ++e) acc += sH2[b * 32 + e] * W3v[e * 2 + j];
    out[32 + tid] = acc;  // tid == b*2+j
  }
}

// ---------------------------------------------------------------------------
// Host launch
// ---------------------------------------------------------------------------
extern "C" void kernel_launch(void* const* d_in, const int* in_sizes, int n_in,
                              void* d_out, int out_size, void* d_ws, size_t ws_size,
                              hipStream_t stream) {
  (void)in_sizes; (void)n_in; (void)out_size; (void)ws_size;
  const float* x    = (const float*)d_in[0];
  const float* adj  = (const float*)d_in[1];
  const float* Wemb = (const float*)d_in[2];
  const float* bemb = (const float*)d_in[3];
  const float* W1   = (const float*)d_in[4];
  const float* a1   = (const float*)d_in[5];
  const float* W2   = (const float*)d_in[6];
  const float* a2   = (const float*)d_in[7];
  const float* Wih  = (const float*)d_in[8];
  const float* Whh  = (const float*)d_in[9];
  const float* bih  = (const float*)d_in[10];
  const float* bhh  = (const float*)d_in[11];
  const float* hW1  = (const float*)d_in[12];
  const float* hb1  = (const float*)d_in[13];
  const float* hW2  = (const float*)d_in[14];
  const float* hb2  = (const float*)d_in[15];
  const float* W3d  = (const float*)d_in[16];
  const float* b3d  = (const float*)d_in[17];
  const float* W3r  = (const float*)d_in[18];
  const float* b3r  = (const float*)d_in[19];
  const float* W3v  = (const float*)d_in[20];
  const float* b3v  = (const float*)d_in[21];

  float* out = (float*)d_out;
  float* ws  = (float*)d_ws;
  float* xg_out = ws;                 // 16*2048
  float* xl_out = ws + 32768;         // 16*128
  float* seqA   = ws + 34816;         // 16*96*128
  float* seqB   = ws + 231424;        // 16*96*128

  float* attn2 = out + 64;            // d(16)|r(16)|v(32)|attn2

  gat_fused_kernel<<<dim3(1536), dim3(256), GAT_LDS, stream>>>(
      x, adj, Wemb, bemb, W1, a1, W2, a2, attn2, xg_out);
  lstm_kernel<<<dim3(1), dim3(256), LSTM_LDS, stream>>>(
      x, Wih, Whh, bih, bhh, seqA, seqB, xl_out);
  head_kernel<<<dim3(3), dim3(128), HEAD_LDS, stream>>>(
      xl_out, xg_out, hW1, hb1, hW2, hb2, W3d, b3d, W3r, b3r, W3v, b3v, out);
}
